// SpMiddleResNetFHDFocal_32298154066281
// MI455X (gfx1250) — matrix operand order repaired and checked by reading, not compiled, not run on hardware
//
#include <hip/hip_runtime.h>

typedef __attribute__((ext_vector_type(16))) _Float16 v16h;
typedef __attribute__((ext_vector_type(8)))  _Float16 v8h;
typedef __attribute__((ext_vector_type(8)))  float    v8f;
typedef __attribute__((ext_vector_type(4)))  float    v4f;

#define DD 16
#define HH 112
#define WW 112
#define NVOX (DD*HH*WW)        // 200704
#define HWPLANE (HH*WW)        // 12544
#define DP (DD+2)
#define HP (HH+2)
#define WP (WW+2)
#define NPVOX (DP*HP*WP)       // 233928

__global__ void zero_f16(_Float16* __restrict__ p, int n8) {
    int i = blockIdx.x * 256 + threadIdx.x;
    if (i < n8) { v8h z = {}; ((v8h*)p)[i] = z; }
}

__global__ void x_to_f16(const float* __restrict__ x, _Float16* __restrict__ feat) {
    int idx = blockIdx.x * 256 + threadIdx.x;
    if (idx >= NVOX * 16) return;
    int c = idx & 15;
    int vox = idx >> 4;
    int d = vox / HWPLANE;
    int hw = vox % HWPLANE;
    int h = hw / WW, w = hw % WW;
    size_t pvox = ((size_t)(d + 1) * HP + (h + 1)) * WP + (w + 1);
    feat[pvox * 16 + c] = (_Float16)x[(size_t)c * NVOX + vox];
}

__global__ void pack_w(const float* __restrict__ w, _Float16* __restrict__ pw,
                       int Cin, int Cout, int nchunk) {
    int idx = blockIdx.x * 256 + threadIdx.x;
    int total = nchunk * Cout * 32;
    if (idx >= total) return;
    int k    = idx & 31;
    int cout = (idx >> 5) % Cout;
    int ch   = idx / (32 * Cout);
    int gk   = ch * 32 + k;
    float val = 0.f;
    if (gk < 27 * Cin) {
        int tap = gk / Cin, cin = gk % Cin;
        val = w[((size_t)tap * Cin + cin) * Cout + cout];
    }
    pw[((size_t)ch * Cout + cout) * 32 + k] = (_Float16)val;
}

__global__ void bn_fold(const float* __restrict__ gamma, const float* __restrict__ beta,
                        const float* __restrict__ mean,  const float* __restrict__ var,
                        const float* __restrict__ b,
                        float* __restrict__ scale, float* __restrict__ bias, int C) {
    int i = blockIdx.x * 256 + threadIdx.x;
    if (i >= C) return;
    float s = gamma[i] * rsqrtf(var[i] + 1e-3f);
    float bb = b ? b[i] : 0.f;
    scale[i] = s;
    bias[i]  = s * (bb - mean[i]) + beta[i];
}

__global__ void imp_map(const _Float16* __restrict__ feat, const float* __restrict__ wimp,
                        const float* __restrict__ bimp, float* __restrict__ imp, int Cin) {
    int vox = blockIdx.x * 256 + threadIdx.x;
    if (vox >= NVOX) return;
    int d = vox / HWPLANE;
    int hw = vox % HWPLANE;
    int h = hw / WW, w = hw % WW;
    long pbase = ((long)(d + 1) * HP + (h + 1)) * WP + (w + 1);
    float acc = bimp[0];
    #pragma unroll
    for (int kd = 0; kd < 3; ++kd)
    #pragma unroll
    for (int kh = 0; kh < 3; ++kh)
    #pragma unroll
    for (int kw = 0; kw < 3; ++kw) {
        const int tap  = (kd * 3 + kh) * 3 + kw;
        const long toff = ((long)(kd - 1) * HP + (kh - 1)) * WP + (kw - 1);
        const _Float16* p = feat + (pbase + toff) * Cin;
        const float* wp = wimp + tap * Cin;
        for (int c = 0; c < Cin; c += 8) {
            v8h v = *(const v8h*)(p + c);
            #pragma unroll
            for (int j = 0; j < 8; ++j) acc += (float)v[j] * wp[c + j];
        }
    }
    imp[vox] = 1.0f / (1.0f + __expf(-acc));
}

// ---------------------------------------------------------------------------
// Implicit-GEMM 3x3x3 conv via V_WMMA_F32_16X16X32_F16.
// Block = 128 threads = 4 waves; each wave owns one 16-voxel M-tile, all four
// waves share the same NT x 16-cout N-tiles. Per K-chunk the block stages the
// NT KB B-slab into (double-buffered) LDS once -> 4x less global B traffic.
// A loads stay per-wave global_load_b128 with compile-time immediate offsets
// (padded feature maps, templated CIN/COUT).
// ---------------------------------------------------------------------------
template<int CIN, int COUT, int NT>
__global__ __launch_bounds__(128)
void conv3d_wmma(const _Float16* __restrict__ feat,    // padded [DP,HP,WP,CIN]
                 const _Float16* __restrict__ wpack,   // [nchunk][COUT][32]
                 const float* __restrict__ scale,
                 const float* __restrict__ bias,
                 const float* __restrict__ imp,        // [NVOX] or null
                 const _Float16* __restrict__ res16,   // padded [NPVOX][COUT] or null
                 const float* __restrict__ res32,      // [COUT][NVOX] or null
                 _Float16* __restrict__ out16,         // padded [NPVOX][COUT] or null
                 float* __restrict__ out32,            // [COUT][NVOX] or null
                 int relu) {
    constexpr int CPT = (CIN >= 32) ? CIN / 32 : 1;    // chunks per tap
    constexpr int NCH = (CIN >= 32) ? 27 * CPT : 14;   // total K chunks
    constexpr int SLAB = NT * 512;                     // halves per B slab

    __shared__ _Float16 lds[2][SLAB];

    const int tid  = threadIdx.x;
    const int lane = tid & 31;
    const int wid  = tid >> 5;

    const int tile = blockIdx.x * 4 + wid;             // per-wave M tile
    const int vox0 = tile * 16;
    const int d  = vox0 / HWPLANE;
    const int hw = vox0 % HWPLANE;
    const int h  = hw / WW;
    const int w0 = hw % WW;

    const int am    = lane & 15;
    const int aOff  = (lane < 16) ? 0 : 8;
    const int bKoff = (lane < 16) ? 0 : 8;
    const int cout0 = blockIdx.y * (16 * NT) + (lane & 15);

    const _Float16* abase = feat +
        (((long)(d + 1) * HP + (h + 1)) * WP + (w0 + 1) + am) * CIN + aOff;
    // block's B region: couts [blockIdx.y*16NT, +16NT) are contiguous 32-half rows
    const _Float16* gB = wpack + (size_t)blockIdx.y * (16 * NT) * 32;
    // per-lane LDS read base: cout_local = (lane&15) + t*16 -> offset +t*512
    const _Float16* lbase0 = &lds[0][(lane & 15) * 32 + bKoff];
    const _Float16* lbase1 = &lds[1][(lane & 15) * 32 + bKoff];

    auto stage = [&](int ch, int buf) {                // 128 thr x 16B x NT/2
        const _Float16* src = gB + (size_t)ch * COUT * 32;
        #pragma unroll
        for (int r = 0; r < NT / 2; ++r) {
            int o = r * 1024 + tid * 8;
            *(v8h*)(&lds[buf][o]) = *(const v8h*)(src + o);
        }
    };

    v8f acc[NT] = {};
    union AFrag { v16h v; v8h seg[2]; };

    stage(0, 0);
    __syncthreads();

    #pragma unroll
    for (int ch = 0; ch < NCH; ++ch) {
        if (ch + 1 < NCH) stage(ch + 1, (ch + 1) & 1);

        // ---- A fragment (constant immediate offsets after unroll)
        AFrag a;
        if (CIN >= 32) {
            const int tap = ch / CPT, cc = ch % CPT;
            const int kd = tap / 9, kh = (tap / 3) % 3, kw = tap % 3;
            const int toff = (((kd - 1) * HP + (kh - 1)) * WP + (kw - 1)) * CIN;
            a.seg[0] = *(const v8h*)(abase + toff + cc * 32);
            a.seg[1] = *(const v8h*)(abase + toff + cc * 32 + 16);
        } else {
            #pragma unroll
            for (int s = 0; s < 2; ++s) {
                const int tap = 2 * ch + s;
                if (tap < 27) {
                    const int kd = tap / 9, kh = (tap / 3) % 3, kw = tap % 3;
                    const int toff = (((kd - 1) * HP + (kh - 1)) * WP + (kw - 1)) * 16;
                    a.seg[s] = *(const v8h*)(abase + toff);
                } else {
                    v8h z = {}; a.seg[s] = z;
                }
            }
        }

        // ---- B fragments from LDS, NT wmma
        const _Float16* lp = (ch & 1) ? lbase1 : lbase0;
        #pragma unroll
        for (int t = 0; t < NT; ++t) {
            v8h b0 = *(const v8h*)(lp + t * 512);      // 2x ds_load 16B: k = 8h+s, 16+8h+s
            v8h b1 = *(const v8h*)(lp + t * 512 + 16);
            v16h b = __builtin_shufflevector(b0, b1, 0, 1, 2, 3, 4, 5, 6, 7, 8, 9, 10, 11, 12, 13, 14, 15);
            acc[t] = __builtin_amdgcn_wmma_f32_16x16x32_f16(
                false, a.v, false, b, (short)0, acc[t], false, false);
        }
        __syncthreads();
    }

    // ---- epilogue: BN, focal gate, residual, ReLU
    const int mbase = (lane < 16) ? 0 : 8;             // C layout: VGPR v -> M = v+mbase
    const long prow = ((long)(d + 1) * HP + (h + 1)) * WP + 1;   // + w -> padded idx
    #pragma unroll
    for (int t = 0; t < NT; ++t) {
        const int cout = cout0 + t * 16;
        const float sc = scale[cout];
        const float bi = bias[cout];
        float y[8];
        #pragma unroll
        for (int v = 0; v < 8; ++v) {
            int vox = vox0 + mbase + v;
            int w = w0 + mbase + v;
            float r = acc[t][v] * sc + bi;
            if (imp) { float g = imp[vox]; r = (g > 0.5f) ? r * g : 0.f; }
            if (res16) r += (float)res16[(size_t)(prow + w) * COUT + cout];
            if (res32) r += res32[(size_t)cout * NVOX + vox];
            if (relu) r = fmaxf(r, 0.f);
            y[v] = r;
        }
        if (out16) {
            #pragma unroll
            for (int v = 0; v < 8; ++v) {
                int w = w0 + mbase + v;
                out16[(size_t)(prow + w) * COUT + cout] = (_Float16)y[v];
            }
        }
        if (out32) {
            float* op = out32 + (size_t)cout * NVOX + vox0 + mbase;
            *(v4f*)op       = v4f{y[0], y[1], y[2], y[3]};
            *(v4f*)(op + 4) = v4f{y[4], y[5], y[6], y[7]};
        }
    }
}

struct Focal { int w, b, gamma, beta, mean, var, wimp, bimp; };
static Focal focal_at(int base) {
    return Focal{base, base + 1, base + 2, base + 3, base + 4, base + 5, base + 6, base + 7};
}

extern "C" void kernel_launch(void* const* d_in, const int* in_sizes, int n_in,
                              void* d_out, int out_size, void* d_ws, size_t ws_size,
                              hipStream_t stream) {
    (void)in_sizes; (void)n_in; (void)out_size; (void)ws_size;
    char* ws = (char*)d_ws;
    size_t off = 0;
    auto alloc = [&](size_t bytes) -> void* {
        off = (off + 255) & ~(size_t)255;
        void* p = ws + off;
        off += bytes;
        return p;
    };

    _Float16* feat0 = (_Float16*)alloc((size_t)NPVOX * 16  * 2);
    _Float16* h0    = (_Float16*)alloc((size_t)NPVOX * 32  * 2);
    _Float16* t1    = (_Float16*)alloc((size_t)NPVOX * 32  * 2);
    _Float16* t2    = (_Float16*)alloc((size_t)NPVOX * 32  * 2);
    _Float16* h1    = (_Float16*)alloc((size_t)NPVOX * 128 * 2);
    _Float16* hds1p = (_Float16*)alloc((size_t)NPVOX * 128 * 2);  // padded residual
    float*    impA  = (float*)   alloc((size_t)NVOX * 4);
    float*    impB  = (float*)   alloc((size_t)NVOX * 4);

    const float* x = (const float*)d_in[0];
    float* out = (float*)d_out;

    auto zero16 = [&](_Float16* p, size_t elems) {
        int n8 = (int)(elems / 8);
        zero_f16<<<(n8 + 255) / 256, 256, 0, stream>>>(p, n8);
    };
    zero16(feat0, (size_t)NPVOX * 16);
    zero16(h0,    (size_t)NPVOX * 32);
    zero16(t1,    (size_t)NPVOX * 32);
    zero16(t2,    (size_t)NPVOX * 32);
    zero16(h1,    (size_t)NPVOX * 128);

    x_to_f16<<<(NVOX * 16 + 255) / 256, 256, 0, stream>>>(x, feat0);

    struct Prep { _Float16* pw; float* scale; float* bias; int Cin, Cout; };
    auto prepare = [&](int w_idx, int b_idx, int g_idx, int be_idx, int m_idx, int v_idx,
                       int Cin, int Cout) -> Prep {
        int nchunk = (27 * Cin + 31) / 32;
        Prep p;
        p.pw    = (_Float16*)alloc((size_t)nchunk * Cout * 32 * 2);
        p.scale = (float*)alloc(Cout * 4);
        p.bias  = (float*)alloc(Cout * 4);
        p.Cin = Cin; p.Cout = Cout;
        int total = nchunk * Cout * 32;
        pack_w<<<(total + 255) / 256, 256, 0, stream>>>(
            (const float*)d_in[w_idx], p.pw, Cin, Cout, nchunk);
        bn_fold<<<(Cout + 255) / 256, 256, 0, stream>>>(
            (const float*)d_in[g_idx], (const float*)d_in[be_idx],
            (const float*)d_in[m_idx], (const float*)d_in[v_idx],
            b_idx >= 0 ? (const float*)d_in[b_idx] : nullptr,
            p.scale, p.bias, Cout);
        return p;
    };

    auto run_conv = [&](const Prep& p, const _Float16* fin,
                        const float* gate, const _Float16* r16, const float* r32,
                        _Float16* o16, float* o32, int relu) {
        #define LC(CI, CO, NTT) do { dim3 grid(NVOX / 64, (CO) / (16 * (NTT)));       \
            conv3d_wmma<CI, CO, NTT><<<grid, 128, 0, stream>>>(                       \
                fin, p.pw, p.scale, p.bias, gate, r16, r32, o16, o32, relu); } while (0)
        if      (p.Cin == 16  && p.Cout == 32 ) LC(16,  32,  2);
        else if (p.Cin == 32  && p.Cout == 32 ) LC(32,  32,  2);
        else if (p.Cin == 32  && p.Cout == 128) LC(32,  128, 4);
        else if (p.Cin == 128 && p.Cout == 32 ) LC(128, 32,  2);
        else if (p.Cin == 32  && p.Cout == 512) LC(32,  512, 4);
        else if (p.Cin == 128 && p.Cout == 512) LC(128, 512, 4);
        #undef LC
    };

    auto run_imp = [&](const Focal& f, const _Float16* fin, int Cin, float* impout) {
        imp_map<<<(NVOX + 255) / 256, 256, 0, stream>>>(
            fin, (const float*)d_in[f.wimp], (const float*)d_in[f.bimp], impout, Cin);
    };

    Focal b1c1 = focal_at(6),  b1c2 = focal_at(14), b1c3 = focal_at(22), b1ds = focal_at(30);
    Focal b2c1 = focal_at(38), b2c2 = focal_at(46), b2c3 = focal_at(54), b2ds = focal_at(62);

    // conv_in: 16->32, BN, ReLU
    Prep Pin = prepare(1, -1, 2, 3, 4, 5, 16, 32);
    run_conv(Pin, feat0, nullptr, nullptr, nullptr, h0, nullptr, 1);

    // block1 (32 -> 128)
    Prep P11 = prepare(b1c1.w, b1c1.b, b1c1.gamma, b1c1.beta, b1c1.mean, b1c1.var, 32, 32);
    Prep P12 = prepare(b1c2.w, b1c2.b, b1c2.gamma, b1c2.beta, b1c2.mean, b1c2.var, 32, 32);
    Prep P13 = prepare(b1c3.w, b1c3.b, b1c3.gamma, b1c3.beta, b1c3.mean, b1c3.var, 32, 128);
    Prep P1d = prepare(b1ds.w, b1ds.b, b1ds.gamma, b1ds.beta, b1ds.mean, b1ds.var, 32, 128);

    run_imp(b1c1, h0, 32, impA);
    run_conv(P11, h0, impA, nullptr, nullptr, t1, nullptr, 1);
    run_imp(b1c2, t1, 32, impB);
    run_conv(P12, t1, impB, nullptr, nullptr, t2, nullptr, 0);
    run_imp(b1ds, h0, 32, impB);
    run_conv(P1d, h0, impB, nullptr, nullptr, hds1p, nullptr, 0);   // ds (padded f16)
    run_imp(b1c3, t2, 32, impA);
    run_conv(P13, t2, impA, hds1p, nullptr, h1, nullptr, 1);        // relu(c3 + ds)

    // block2 (128 -> 512) -> NCDHW f32 d_out
    Prep P21 = prepare(b2c1.w, b2c1.b, b2c1.gamma, b2c1.beta, b2c1.mean, b2c1.var, 128, 32);
    Prep P22 = prepare(b2c2.w, b2c2.b, b2c2.gamma, b2c2.beta, b2c2.mean, b2c2.var, 32, 32);
    Prep P23 = prepare(b2c3.w, b2c3.b, b2c3.gamma, b2c3.beta, b2c3.mean, b2c3.var, 32, 512);
    Prep P2d = prepare(b2ds.w, b2ds.b, b2ds.gamma, b2ds.beta, b2ds.mean, b2ds.var, 128, 512);

    run_imp(b2c1, h1, 128, impA);
    run_conv(P21, h1, impA, nullptr, nullptr, t1, nullptr, 1);
    run_imp(b2c2, t1, 32, impB);
    run_conv(P22, t1, impB, nullptr, nullptr, t2, nullptr, 0);
    run_imp(b2ds, h1, 128, impB);
    run_conv(P2d, h1, impB, nullptr, nullptr, nullptr, out, 0);     // ds -> d_out
    run_imp(b2c3, t2, 32, impA);
    run_conv(P23, t2, impA, nullptr, out, nullptr, out, 1);         // relu(c3 + d_out)
}
